// GRUCell_59227599012282
// MI455X (gfx1250) — matrix operand order repaired and checked by reading, not compiled, not run on hardware
//
#include <hip/hip_runtime.h>
#include <hip/hip_bf16.h>
#include <stdint.h>

typedef __attribute__((ext_vector_type(16))) __bf16 v16bf;
typedef __attribute__((ext_vector_type(8)))  __bf16 v8bf;
typedef __attribute__((ext_vector_type(8)))  float  v8f;

typedef unsigned short bf16_t;   // raw bf16 bits

#define B_DIM  8192
#define H_DIM  2048
#define TM     128
#define TN     128
#define TK     64
#define LDSK   72    // padded LDS row stride (elements); 72*2B = 144B, multiple of 16B

__device__ __forceinline__ bf16_t f2bf(float f) {
  unsigned u = __float_as_uint(f);
  u += 0x7FFFu + ((u >> 16) & 1u);       // round-to-nearest-even
  return (bf16_t)(u >> 16);
}

// ---- pack kernels -----------------------------------------------------------

__global__ __launch_bounds__(256) void pack_x_h(const float* __restrict__ x,
                                                const float* __restrict__ h,
                                                bf16_t* __restrict__ out) {
  long i = (long)blockIdx.x * blockDim.x + threadIdx.x;   // over 8192*4096
  long row = i >> 12;
  long col = i & 4095;
  float v = (col < 2048) ? x[row * 2048 + col] : h[row * 2048 + (col - 2048)];
  out[i] = f2bf(v);
}

__global__ __launch_bounds__(256) void pack_wt2(const float* __restrict__ W,
                                                const float* __restrict__ U,
                                                bf16_t* __restrict__ outT) {
  // outT: [N=2048][K=4096], outT[n*4096+k] = (k<2048 ? W[k][n] : U[k-2048][n])
  long i = (long)blockIdx.x * blockDim.x + threadIdx.x;
  long n = i >> 12;
  long k = i & 4095;
  float v = (k < 2048) ? W[k * 2048 + n] : U[(k - 2048) * 2048 + n];
  outT[i] = f2bf(v);
}

__global__ __launch_bounds__(256) void pack_wt1(const float* __restrict__ W,
                                                bf16_t* __restrict__ outT) {
  // outT: [N=2048][K=2048], outT[n*2048+k] = W[k][n]
  long i = (long)blockIdx.x * blockDim.x + threadIdx.x;
  long n = i >> 11;
  long k = i & 2047;
  outT[i] = f2bf(W[k * 2048 + n]);
}

// ---- async global -> LDS tile staging (CDNA5 async path, ASYNCcnt) ----------

// 128 rows x 64 cols bf16 tile; 256 threads: 2 threads/row, 64 contiguous bytes each.
// Four global_load_async_to_lds_b128; INST_OFFSET applies to both LDS and global side.
__device__ __forceinline__ void async_tile(bf16_t* __restrict__ dstLds,
                                           const bf16_t* __restrict__ src, long ld) {
  const int t   = threadIdx.x;
  const int row = t >> 1;
  const int seg = (t & 1) * 32;                       // element offset within row
  const bf16_t* g = src + (long)row * ld + seg;
  const uint32_t l = (uint32_t)(uintptr_t)(dstLds + row * LDSK + seg); // LDS byte addr = low 32 bits
  asm volatile(
      "global_load_async_to_lds_b128 %0, %1, off\n\t"
      "global_load_async_to_lds_b128 %0, %1, off offset:16\n\t"
      "global_load_async_to_lds_b128 %0, %1, off offset:32\n\t"
      "global_load_async_to_lds_b128 %0, %1, off offset:48"
      :: "v"(l), "v"((unsigned long long)(uintptr_t)g)
      : "memory");
}

// ---- LDS fragment loads (ISA 7.12.2 16-bit layouts) -------------------------

__device__ __forceinline__ v16bf load_a_frag(const bf16_t* As, int mBase, int kk,
                                             int half, int lrow) {
  // lane half 0: K {0..7, 16..23}; half 1: K {8..15, 24..31} (relative to kk)
  union { v16bf v; v8bf h[2]; } u;
  const bf16_t* p = As + (mBase + lrow) * LDSK + kk;
  u.h[0] = *reinterpret_cast<const v8bf*>(p + half * 8);
  u.h[1] = *reinterpret_cast<const v8bf*>(p + 16 + half * 8);
  return u.v;
}

__device__ __forceinline__ v16bf load_b_frag(const bf16_t* Bs, int nBase, int kk,
                                             int half, int lrow) {
  // lane half 0: K {0..7, 16..23} of column N=lrow; half 1: K {8..15, 24..31}
  union { v16bf v; v8bf h[2]; } u;
  const bf16_t* p = Bs + (nBase + lrow) * LDSK + kk + half * 8;
  u.h[0] = *reinterpret_cast<const v8bf*>(p);
  u.h[1] = *reinterpret_cast<const v8bf*>(p + 16);
  return u.v;
}

// EPI: 0 = z = sigmoid(acc+b)              -> outF
//      1 = rh = bf16(sigmoid(acc+b) * h)   -> outBf
//      2 = cpre = acc + b                  -> outF
//      3 = h_t = (1-z)h + z*tanh(cpre+acc) -> outF
template <int EPI>
__global__ __launch_bounds__(256)
void gemm_bf16(const bf16_t* __restrict__ A, long lda,
               const bf16_t* __restrict__ Bt, long ldb, int K,
               const float* __restrict__ bias,
               const float* __restrict__ h_prev,
               const float* __restrict__ zbuf,
               const float* __restrict__ cpre,
               float* __restrict__ outF,
               bf16_t* __restrict__ outBf) {
  __shared__ bf16_t As[2][TM * LDSK];
  __shared__ bf16_t Bs[2][TN * LDSK];

  const int rowBase = blockIdx.y * TM;
  const int colBase = blockIdx.x * TN;

  const int wave  = threadIdx.x >> 5;
  const int lane  = threadIdx.x & 31;
  const int half  = lane >> 4;
  const int lrow  = lane & 15;
  const int waveM = wave & 3;    // 4 waves along M: 32 rows each
  const int waveN = wave >> 2;   // 2 waves along N: 64 cols each

  v8f acc[2][4];
#pragma unroll
  for (int i = 0; i < 2; ++i)
#pragma unroll
    for (int j = 0; j < 4; ++j) acc[i][j] = v8f{0.f, 0.f, 0.f, 0.f, 0.f, 0.f, 0.f, 0.f};

  const bf16_t* aG = A + (long)rowBase * lda;
  const bf16_t* bG = Bt + (long)colBase * ldb;

  const int nIter = K / TK;

  // prologue: stage tile 0 and tile 1 (8 async loads per tile-pair per thread)
  async_tile(&As[0][0], aG, lda);
  async_tile(&Bs[0][0], bG, ldb);
  if (nIter > 1) {
    async_tile(&As[1][0], aG + TK, lda);
    async_tile(&Bs[1][0], bG + TK, ldb);
  }

  for (int i = 0; i < nIter; ++i) {
    // wait until current tile's 8 async loads have landed in LDS
    if (i + 1 < nIter) asm volatile("s_wait_asynccnt 8" ::: "memory");
    else               asm volatile("s_wait_asynccnt 0" ::: "memory");
    __syncthreads();

    const int cur = i & 1;
    const bf16_t* Asb = &As[cur][0];
    const bf16_t* Bsb = &Bs[cur][0];
#pragma unroll
    for (int kk = 0; kk < TK; kk += 32) {
      v16bf af[2], bfv[4];
#pragma unroll
      for (int mt = 0; mt < 2; ++mt)
        af[mt] = load_a_frag(Asb, waveM * 32 + mt * 16, kk, half, lrow);
#pragma unroll
      for (int nt = 0; nt < 4; ++nt)
        bfv[nt] = load_b_frag(Bsb, waveN * 64 + nt * 16, kk, half, lrow);
#pragma unroll
      for (int mt = 0; mt < 2; ++mt)
#pragma unroll
        for (int nt = 0; nt < 4; ++nt)
          acc[mt][nt] = __builtin_amdgcn_wmma_f32_16x16x32_bf16(
              false, af[mt], false, bfv[nt], (short)0, acc[mt][nt], false, false);
    }
    __syncthreads();   // all waves done reading buf[cur]

    if (i + 2 < nIter) {  // stage tile i+2 into the buffer just freed
      const long k2 = (long)(i + 2) * TK;
      async_tile(&As[cur][0], aG + k2, lda);
      async_tile(&Bs[cur][0], bG + k2, ldb);
    }
  }

  // epilogue: C layout = VGPR r -> M = r + 8*half, N = lane&15 (ISA 7.12.2)
#pragma unroll
  for (int mt = 0; mt < 2; ++mt) {
#pragma unroll
    for (int nt = 0; nt < 4; ++nt) {
      const int tileRow = rowBase + waveM * 32 + mt * 16;
      const int col     = colBase + waveN * 64 + nt * 16 + lrow;
      float bv = 0.f;
      if constexpr (EPI != 3) bv = bias[col];
#pragma unroll
      for (int r = 0; r < 8; ++r) {
        const int  row = tileRow + r + half * 8;
        const long idx = (long)row * H_DIM + col;
        const float v  = acc[mt][nt][r];
        if constexpr (EPI == 0) {
          outF[idx] = 1.f / (1.f + __expf(-(v + bv)));
        } else if constexpr (EPI == 1) {
          const float rg = 1.f / (1.f + __expf(-(v + bv)));
          outBf[idx] = f2bf(rg * h_prev[idx]);
        } else if constexpr (EPI == 2) {
          outF[idx] = v + bv;
        } else {
          const float z  = zbuf[idx];
          const float ht = tanhf(cpre[idx] + v);
          outF[idx] = (1.f - z) * h_prev[idx] + z * ht;
        }
      }
    }
  }
}

// ---- launch -----------------------------------------------------------------

extern "C" void kernel_launch(void* const* d_in, const int* in_sizes, int n_in,
                              void* d_out, int out_size, void* d_ws, size_t ws_size,
                              hipStream_t stream) {
  const float* x  = (const float*)d_in[0];
  const float* h  = (const float*)d_in[1];
  const float* Wz = (const float*)d_in[2];
  const float* Wr = (const float*)d_in[3];
  const float* Wh = (const float*)d_in[4];
  const float* Uz = (const float*)d_in[5];
  const float* Ur = (const float*)d_in[6];
  const float* Uh = (const float*)d_in[7];
  const float* bz = (const float*)d_in[8];
  const float* br = (const float*)d_in[9];
  const float* bh = (const float*)d_in[10];
  float* out = (float*)d_out;

  char* ws = (char*)d_ws;
  bf16_t* Abuf = (bf16_t*)(ws);                    // [8192][4096] bf16   64 MB
  bf16_t* WzuT = (bf16_t*)(ws + (64ll  << 20));    // [2048][4096] bf16   16 MB
  bf16_t* WruT = (bf16_t*)(ws + (80ll  << 20));    // [2048][4096] bf16   16 MB
  bf16_t* WhT  = (bf16_t*)(ws + (96ll  << 20));    // [2048][2048] bf16    8 MB
  bf16_t* UhT  = (bf16_t*)(ws + (104ll << 20));    // [2048][2048] bf16    8 MB
  float*  zbuf = (float*) (ws + (112ll << 20));    // [8192][2048] f32    64 MB
  float*  cpre = (float*) (ws + (176ll << 20));    // [8192][2048] f32    64 MB
  bf16_t* rh   = (bf16_t*)(ws + (240ll << 20));    // [8192][2048] bf16   32 MB

  pack_x_h<<<(B_DIM * 4096) / 256, 256, 0, stream>>>(x, h, Abuf);
  pack_wt2<<<(2048 * 4096) / 256, 256, 0, stream>>>(Wz, Uz, WzuT);
  pack_wt2<<<(2048 * 4096) / 256, 256, 0, stream>>>(Wr, Ur, WruT);
  pack_wt1<<<(2048 * 2048) / 256, 256, 0, stream>>>(Wh, WhT);
  pack_wt1<<<(2048 * 2048) / 256, 256, 0, stream>>>(Uh, UhT);

  dim3 grid(H_DIM / TN, B_DIM / TM);   // 16 x 64 blocks
  // z = sigmoid([x|h]·[Wz;Uz] + bz)
  gemm_bf16<0><<<grid, 256, 0, stream>>>(Abuf, 4096, WzuT, 4096, 4096,
                                         bz, nullptr, nullptr, nullptr, zbuf, nullptr);
  // rh = bf16(sigmoid([x|h]·[Wr;Ur] + br) * h)
  gemm_bf16<1><<<grid, 256, 0, stream>>>(Abuf, 4096, WruT, 4096, 4096,
                                         br, h, nullptr, nullptr, nullptr, rh);
  // cpre = x·Wh + bh   (uses x-half of Abuf: lda=4096, K=2048)
  gemm_bf16<2><<<grid, 256, 0, stream>>>(Abuf, 4096, WhT, 2048, 2048,
                                         bh, nullptr, nullptr, nullptr, cpre, nullptr);
  // h_t = (1-z)*h + z*tanh(cpre + rh·Uh)
  gemm_bf16<3><<<grid, 256, 0, stream>>>(rh, 2048, UhT, 2048, 2048,
                                         nullptr, h, zbuf, cpre, out, nullptr);
}
